// GatedEquivariantBlock_60919816127126
// MI455X (gfx1250) — compile-verified, matrix operand order checked by reading, not run on hardware
//
#include <hip/hip_runtime.h>
#include <hip/hip_bf16.h>
#include <math.h>

typedef __attribute__((ext_vector_type(16))) _Float16 v16h;
typedef __attribute__((ext_vector_type(8)))  _Float16 v8h;
typedef __attribute__((ext_vector_type(8)))  float    v8f;

#define NPATH 11
#define NENT  83
#define NCHUNKS_TOTAL 216   // 3456/16

// Sparse real Wigner-3j tables (unit Frobenius norm), e3nn real-SH basis.
// constexpr (not __constant__) so the compiler folds entries into immediates
// inside the fully-unrolled per-path code.
#define WA 0.31622776f   // 1/sqrt(10)
#define WB 0.18257419f   // 1/sqrt(30)
#define WC 0.36514837f   // 2/sqrt(30)
#define WD 0.20702060f   // sqrt(3/70)
#define WE 0.23904572f   // sqrt(2/35)
#define WF 0.11952286f   // sqrt(1/70)
constexpr int cg_i[NENT] = {
  0,
  0,0,0,
  0,0,0,0,0,
  0,1,2,
  0,1,2,
  2,0,0,1,2,1,2,0,0,2,1,
  2,0,0,1,2,1,2,0,0,2,1,
  0,1,2,3,4,
  0,0,1,1,3,3,4,4,2,2,2,
  0,1,2,3,4,
  0,0,1,1,3,3, 0,0,2,4,4,2, 3,3,2,1,1,2, 2, 3,3,4, 1,1,4
};
constexpr int cg_j[NENT] = {
  0,
  0,1,2,
  0,1,2,3,4,
  0,0,0,
  0,1,2,
  0,2,1,0,1,2,2,0,0,2,1,
  0,0,1,1,3,3,4,4,2,2,2,
  0,0,0,0,0,
  2,0,0,1,2,1,2,0,0,2,1,
  0,1,2,3,4,
  1,3,0,3,0,1, 0,2,0,4,2,4, 3,2,3,1,2,1, 2, 3,4,3, 1,4,1
};
constexpr int cg_k[NENT] = {
  0,
  0,1,2,
  0,1,2,3,4,
  0,1,2,
  0,0,0,
  0,0,1,1,3,3,4,4,2,2,2,
  0,2,1,0,1,2,2,0,0,2,1,
  0,1,2,3,4,
  0,2,1,0,1,2,2,0,0,2,1,
  0,0,0,0,0,
  3,1,3,0,1,0, 2,0,0,2,4,4, 2,3,3,2,1,1, 2, 4,3,3, 4,1,1
};
constexpr float cg_v[NENT] = {
  1.0f,
  0.57735027f,0.57735027f,0.57735027f,
  0.44721360f,0.44721360f,0.44721360f,0.44721360f,0.44721360f,
  0.57735027f,0.57735027f,0.57735027f,
  0.57735027f,0.57735027f,0.57735027f,
  WA,WA,WA,WA,WA,WA,WA,-WA,-WB,-WB,WC,
  WA,WA,WA,WA,WA,WA,WA,-WA,-WB,-WB,WC,
  0.44721360f,0.44721360f,0.44721360f,0.44721360f,0.44721360f,
  WA,WA,WA,WA,WA,WA,WA,-WA,-WB,-WB,WC,
  0.44721360f,0.44721360f,0.44721360f,0.44721360f,0.44721360f,
  WD,WD,WD,WD,WD,WD, -WE,-WE,-WE,-WE,-WE,-WE, WF,WF,WF,WF,WF,WF, WE, WD,WD,WD, -WD,-WD,-WD
};

// ---------------- Kernel 1: radial MLP (fp32) -> h2 in f16 ----------------
__global__ __launch_bounds__(256) void mlp_kernel(
    const float* __restrict__ radial, const float* __restrict__ W1,
    const float* __restrict__ b1, const float* __restrict__ W2,
    const float* __restrict__ b2, _Float16* __restrict__ h2, int E)
{
  __shared__ float s_h1[4][64];
  const int le = threadIdx.x >> 6;   // local edge 0..3
  const int j  = threadIdx.x & 63;   // hidden unit
  const int e  = blockIdx.x * 4 + le;
  float h1 = 0.f;
  if (e < E) {
    const float* r = radial + (size_t)e * 8;
    float acc = b1[j];
    #pragma unroll
    for (int i = 0; i < 8; ++i) acc += r[i] * W1[i*64 + j];
    h1 = acc / (1.f + __expf(-acc));   // silu
  }
  s_h1[le][j] = h1;
  __syncthreads();
  if (e < E) {
    float acc = b2[j];
    #pragma unroll 8
    for (int c = 0; c < 64; ++c) acc += s_h1[le][c] * W2[c*64 + j];
    float h = acc / (1.f + __expf(-acc));
    h2[(size_t)e*64 + j] = (_Float16)h;
  }
}

// ---------------- Kernel 2: pack W3 into f16 WMMA B-fragment order ----------------
// layout: [chunk cc][kstep s][lane l][t 0..15] ; lane l holds column cc*16+(l&15),
// K = s*32 + ((l>=16)?8:0) + (t<8 ? t : t+8)  (mirror of documented 16-bit A layout)
__global__ __launch_bounds__(256) void pack_w3_kernel(
    const float* __restrict__ W3, _Float16* __restrict__ w3p)
{
  const int idx = blockIdx.x * 256 + threadIdx.x;
  if (idx >= NCHUNKS_TOTAL * 2 * 32 * 16) return;
  const int t  = idx & 15;
  const int l  = (idx >> 4) & 31;
  const int s  = (idx >> 9) & 1;
  const int cc = idx >> 10;
  const int col = cc * 16 + (l & 15);
  const int k   = s * 32 + ((l >= 16) ? 8 : 0) + ((t < 8) ? t : t + 8);
  w3p[idx] = (_Float16)W3[(size_t)k * 3456 + col];
}

// ---------------- Kernel 3 helpers ----------------
__device__ __forceinline__ void load_bfrag(const _Float16* __restrict__ w3p,
                                           int cc, int lane, v16h& b0, v16h& b1)
{
  const _Float16* bptr = w3p + ((size_t)(cc * 2) * 32 + lane) * 16;
  v8h blo0 = *(const v8h*)(bptr);
  v8h bhi0 = *(const v8h*)(bptr + 8);
  v8h blo1 = *(const v8h*)(bptr + 512);
  v8h bhi1 = *(const v8h*)(bptr + 520);
  #pragma unroll
  for (int i = 0; i < 8; ++i) {
    b0[i] = blo0[i]; b0[8+i] = bhi0[i];
    b1[i] = blo1[i]; b1[8+i] = bhi1[i];
  }
}

// One tensor-product path, all shape params compile-time.
//   tmp[e][u][k] = sum_entries x[e,u,i]*sh[e,j]*C[i,j,k]
//   wp (16 edges x M1*M3) = h2 @ W3_path + b3, streamed in 16-col WMMA chunks
//   acc[e][OFF3 + w*D3 + k] += coef * wp[e,u,w] * tmp[e,u,k]
template<int M1, int M3LOG, int D1, int D3, int OFF1, int OFF3, int SOFF2,
         int WOFF, int ES, int EE>
__device__ __forceinline__ void run_path(
    bool active, int lane, const v16h& a0, const v16h& a1,
    const float* __restrict__ s_x, const float* __restrict__ s_sh,
    float* __restrict__ s_tmp, float* __restrict__ s_acc,
    const _Float16* __restrict__ w3p, const float* __restrict__ b3, float coef)
{
  // ---- tmp ----
  if (active) {
    constexpr int TOT = 16 * M1 * D3;
    for (int idx = lane; idx < TOT; idx += 32) {
      const int e = idx / (M1 * D3);
      const int r = idx - e * (M1 * D3);
      const int u = r / D3, k = r - u * D3;
      float s = 0.f;
      #pragma unroll
      for (int t = ES; t < EE; ++t) {
        if (cg_k[t] == k)
          s += cg_v[t] * s_x[e*120 + OFF1 + u*D1 + cg_i[t]]
                       * s_sh[e*12 + SOFF2 + cg_j[t]];
      }
      s_tmp[e*160 + u*D3 + k] = s;
    }
  }
  __syncthreads();

  // ---- WMMA chunks, software-pipelined B/bias prefetch ----
  constexpr int NCHUNK = (M1 << M3LOG) >> 4;
  constexpr int CBASE  = WOFF >> 4;
  v16h b0 = {}, b1 = {};
  float bias = 0.f;
  if (active) {
    load_bfrag(w3p, CBASE, lane, b0, b1);
    bias = b3[(CBASE << 4) + (lane & 15)];
  }
  for (int c = 0; c < NCHUNK; ++c) {
    if (active) {   // wave-uniform -> EXEC all-1s inside (WMMA-legal)
      v16h nb0 = {}, nb1 = {};
      float nbias = 0.f;
      if (c + 1 < NCHUNK) {  // prefetch next chunk while this one computes
        load_bfrag(w3p, CBASE + c + 1, lane, nb0, nb1);
        nbias = b3[((CBASE + c + 1) << 4) + (lane & 15)];
      }
      v8f d = {};
      d = __builtin_amdgcn_wmma_f32_16x16x32_f16(false, a0, false, b0, (short)0, d, false, false);
      d = __builtin_amdgcn_wmma_f32_16x16x32_f16(false, a1, false, b1, (short)0, d, false, false);

      // D layout: VGPR r, lane l -> edge = r + 8*(l>=16), col = l&15
      const int pcol = (c << 4) + (lane & 15);
      const int u = pcol >> M3LOG;
      const int w = pcol & ((1 << M3LOG) - 1);
      const int ehi = (lane >= 16) ? 8 : 0;

      // batch-load tmp (one dscnt wait), then fire all LDS atomics
      float tv[8][D3];
      #pragma unroll
      for (int r = 0; r < 8; ++r)
        #pragma unroll
        for (int k = 0; k < D3; ++k)
          tv[r][k] = s_tmp[(r + ehi)*160 + u*D3 + k];
      #pragma unroll
      for (int r = 0; r < 8; ++r) {
        const float cw = coef * (d[r] + bias);
        #pragma unroll
        for (int k = 0; k < D3; ++k)
          atomicAdd(&s_acc[(r + ehi)*120 + OFF3 + w*D3 + k], cw * tv[r][k]);
      }
      b0 = nb0; b1 = nb1; bias = nbias;
    }
  }
  __syncthreads();
}

// ---------------- Kernel 3: fused W3-GEMM (WMMA) + tensor product + scatter ----------------
// one wave (32 lanes) per 16-edge tile, 2 waves per block
__global__ __launch_bounds__(64) void tp_kernel(
    const float* __restrict__ nf,      // N x 120
    const int*   __restrict__ ei,      // 2 x E
    const float* __restrict__ esh,     // E x 9
    const _Float16* __restrict__ h2,   // E x 64 f16
    const _Float16* __restrict__ w3p,  // packed fragments
    const float* __restrict__ b3,      // 3456
    float* __restrict__ agg,           // N x 120 (atomic accum)
    int N, int E)
{
  // per-wave: x 16*120, sh 16*12, tmp 16*160, acc 16*120 => 6592 floats
  __shared__ float smem[2 * 6592];
  const int lane = threadIdx.x & 31;
  const int wave = threadIdx.x >> 5;
  float* s_x   = smem + wave * 6592;
  float* s_sh  = s_x + 16*120;
  float* s_tmp = s_sh + 16*12;
  float* s_acc = s_tmp + 16*160;

  const int tile = blockIdx.x * 2 + wave;
  const int e0 = tile * 16;
  const bool active = (e0 < E);

  // ---- stage gathered node features, sh, zero accumulator ----
  if (active) {
    for (int idx = lane; idx < 16*120; idx += 32) {
      const int e = idx / 120, f = idx - e*120;
      const int ge = e0 + e;
      s_x[idx]  = (ge < E) ? nf[(size_t)ei[ge] * 120 + f] : 0.f;
      s_acc[idx] = 0.f;
    }
    for (int idx = lane; idx < 16*9; idx += 32) {
      const int e = idx / 9, f = idx - e*9;
      const int ge = e0 + e;
      s_sh[e*12 + f] = (ge < E) ? esh[(size_t)ge*9 + f] : 0.f;
    }
  }

  // ---- A fragments: h2 tile (16 edges x 64 hidden), two K=32 steps ----
  // 16-bit A layout: lane<16 -> M=lane, K={0..7,16..23}; lane>=16 -> K={8..15,24..31}
  v16h a0 = {}, a1 = {};
  if (active) {
    int er = e0 + (lane & 15); if (er >= E) er = E - 1;
    const _Float16* row = h2 + (size_t)er * 64;
    const int half = (lane >= 16) ? 8 : 0;
    v8h lo0 = *(const v8h*)(row + half);
    v8h hi0 = *(const v8h*)(row + half + 16);
    v8h lo1 = *(const v8h*)(row + 32 + half);
    v8h hi1 = *(const v8h*)(row + 32 + half + 16);
    #pragma unroll
    for (int i = 0; i < 8; ++i) {
      a0[i] = lo0[i]; a0[8+i] = hi0[i];
      a1[i] = lo1[i]; a1[8+i] = hi1[i];
    }
  }
  __syncthreads();

  const float C0 = 0.13363062f;  // sqrt(1/56)
  const float C1 = 0.20412415f;  // sqrt(3/72)
  const float C2 = 0.27950850f;  // sqrt(5/64)

  //          M1 M3L D1 D3 OFF1 OFF3 SOF2 WOFF  ES EE
  run_path<32, 5, 1, 1,   0,   0, 0,    0,  0,  1>(active, lane, a0, a1, s_x, s_sh, s_tmp, s_acc, w3p, b3, C0);
  run_path<32, 4, 1, 3,   0,  32, 1, 1024,  1,  4>(active, lane, a0, a1, s_x, s_sh, s_tmp, s_acc, w3p, b3, C1);
  run_path<32, 3, 1, 5,   0,  80, 4, 1536,  4,  9>(active, lane, a0, a1, s_x, s_sh, s_tmp, s_acc, w3p, b3, C2);
  run_path<16, 4, 3, 3,  32,  32, 0, 1792,  9, 12>(active, lane, a0, a1, s_x, s_sh, s_tmp, s_acc, w3p, b3, C1);
  run_path<16, 5, 3, 1,  32,   0, 1, 2048, 12, 15>(active, lane, a0, a1, s_x, s_sh, s_tmp, s_acc, w3p, b3, C0);
  run_path<16, 3, 3, 5,  32,  80, 1, 2560, 15, 26>(active, lane, a0, a1, s_x, s_sh, s_tmp, s_acc, w3p, b3, C2);
  run_path<16, 4, 3, 3,  32,  32, 4, 2688, 26, 37>(active, lane, a0, a1, s_x, s_sh, s_tmp, s_acc, w3p, b3, C1);
  run_path< 8, 3, 5, 5,  80,  80, 0, 2944, 37, 42>(active, lane, a0, a1, s_x, s_sh, s_tmp, s_acc, w3p, b3, C2);
  run_path< 8, 4, 5, 3,  80,  32, 1, 3008, 42, 53>(active, lane, a0, a1, s_x, s_sh, s_tmp, s_acc, w3p, b3, C1);
  run_path< 8, 5, 5, 1,  80,   0, 4, 3136, 53, 58>(active, lane, a0, a1, s_x, s_sh, s_tmp, s_acc, w3p, b3, C0);
  run_path< 8, 3, 5, 5,  80,  80, 4, 3392, 58, 83>(active, lane, a0, a1, s_x, s_sh, s_tmp, s_acc, w3p, b3, C2);

  // ---- scatter-add messages to agg[dst] ----
  if (active) {
    for (int idx = lane; idx < 16*120; idx += 32) {
      const int e = idx / 120, f = idx - e*120;
      const int ge = e0 + e;
      if (ge < E) {
        const int dn = ei[E + ge];
        atomicAdd(&agg[(size_t)dn*120 + f], s_acc[e*120 + f]);
      }
    }
  }
}

// ---------------- Kernel 4: per-node block-diagonal linear + residual ----------------
__global__ __launch_bounds__(128) void out_kernel(
    const float* __restrict__ agg, const float* __restrict__ nf,
    const float* __restrict__ Ws0, const float* __restrict__ Ws1,
    const float* __restrict__ Ws2, float* __restrict__ out, int N)
{
  __shared__ float row[120];
  const int n = blockIdx.x;
  const int t = threadIdx.x;
  if (t < 120) row[t] = agg[(size_t)n*120 + t];
  __syncthreads();
  if (t < 120) {
    float y;
    if (t < 32) {
      float s = 0.f;
      #pragma unroll 8
      for (int u = 0; u < 32; ++u) s += row[u] * Ws0[u*32 + t];
      y = s * 0.17677670f;                 // 1/sqrt(32)
    } else if (t < 80) {
      const int r = t - 32, v = r / 3, d = r - 3*v;
      float s = 0.f;
      #pragma unroll
      for (int u = 0; u < 16; ++u) s += row[32 + u*3 + d] * Ws1[u*16 + v];
      y = s * 0.25f;                       // 1/sqrt(16)
    } else {
      const int r = t - 80, v = r / 5, d = r - 5*v;
      float s = 0.f;
      #pragma unroll
      for (int u = 0; u < 8; ++u) s += row[80 + u*5 + d] * Ws2[u*8 + v];
      y = s * 0.35355339f;                 // 1/sqrt(8)
    }
    out[(size_t)n*120 + t] = y + nf[(size_t)n*120 + t];
  }
}

static inline size_t align256(size_t x) { return (x + 255) & ~(size_t)255; }

extern "C" void kernel_launch(void* const* d_in, const int* in_sizes, int n_in,
                              void* d_out, int out_size, void* d_ws, size_t ws_size,
                              hipStream_t stream) {
  const float* nf     = (const float*)d_in[0];   // (N,120)
  const int*   ei     = (const int*)  d_in[1];   // (2,E)
  const float* esh    = (const float*)d_in[2];   // (E,9)
  const float* radial = (const float*)d_in[3];   // (E,8)
  const float* W1     = (const float*)d_in[4];
  const float* b1     = (const float*)d_in[5];
  const float* W2     = (const float*)d_in[6];
  const float* b2     = (const float*)d_in[7];
  const float* W3     = (const float*)d_in[8];   // (64,3456)
  const float* b3     = (const float*)d_in[9];   // (3456,)
  const float* Ws0    = (const float*)d_in[10];
  const float* Ws1    = (const float*)d_in[11];
  const float* Ws2    = (const float*)d_in[12];
  float* out = (float*)d_out;

  const int N = in_sizes[0] / 120;
  const int E = in_sizes[2] / 9;

  // workspace carve-out
  char* ws = (char*)d_ws;
  size_t o = 0;
  _Float16* h2  = (_Float16*)(ws + o); o = align256(o + (size_t)E * 64 * sizeof(_Float16));
  _Float16* w3p = (_Float16*)(ws + o); o = align256(o + (size_t)NCHUNKS_TOTAL * 2 * 32 * 16 * sizeof(_Float16));
  float*    agg = (float*)   (ws + o); o = align256(o + (size_t)N * 120 * sizeof(float));

  hipMemsetAsync(agg, 0, (size_t)N * 120 * sizeof(float), stream);

  mlp_kernel<<<(E + 3) / 4, 256, 0, stream>>>(radial, W1, b1, W2, b2, h2, E);

  const int packElems = NCHUNKS_TOTAL * 2 * 32 * 16;
  pack_w3_kernel<<<(packElems + 255) / 256, 256, 0, stream>>>(W3, w3p);

  const int tiles  = (E + 15) / 16;
  const int blocks = (tiles + 1) / 2;
  tp_kernel<<<blocks, 64, 0, stream>>>(nf, ei, esh, h2, w3p, b3, agg, N, E);

  out_kernel<<<N, 128, 0, stream>>>(agg, nf, Ws0, Ws1, Ws2, out, N);
}
